// QuantumTemporalConvolution_50800873177579
// MI455X (gfx1250) — matrix operand order repaired and checked by reading, not compiled, not run on hardware
//
#include <hip/hip_runtime.h>
#include <hip/hip_bf16.h>
#include <stdint.h>

// Problem constants (match the reference)
#define BATCH   32
#define S_LEN   2048
#define S_PAD   2064        // S_LEN + 16 zero pad rows (time)
#define T_OUT   2033        // S - K + 1
#define CIN     256
#define COUT    256
#define KW      16

// workspace layout:
//   [0, 2MB)            : packed bf16 B-fragments of w_eff, ordered [k][nQuad][ib][nb]
//   [2MB, 2MB+33.8MB)   : padded bf16 x  [BATCH][S_PAD][CIN]
#define WB_HALFS   (KW * (CIN/32) * (COUT/16) * 512)   // 1,048,576 halfs
#define WB_BYTES   (WB_HALFS * 2)                      // 2 MB
#define SLAB_HALFS (32 * 512)                          // per (k,nQuad) slab: 32 frags = 32KB
#define SLAB_BYTES (SLAB_HALFS * 2)

typedef __attribute__((ext_vector_type(16))) __bf16 v16bf;
typedef __attribute__((ext_vector_type(8)))  float  v8f;
typedef __attribute__((ext_vector_type(4)))  unsigned int v4u;
typedef __attribute__((ext_vector_type(8)))  int v8i;
typedef __attribute__((ext_vector_type(4)))  int v4i;

union FragA { v16bf v; uint4 q[2]; };
union FragB { v16bf v; uint4 q[2]; };
union Pack16 { unsigned short s[16]; uint4 q[2]; };

#if defined(__has_builtin)
#if __has_builtin(__builtin_amdgcn_tensor_load_to_lds)
#define HAVE_TDM 1
#endif
#endif

__device__ __forceinline__ unsigned short f2bf(float f) {
  union { float f; uint32_t u; } a; a.f = f;
  uint32_t u = a.u;
  if ((u & 0x7fffffffu) > 0x7f800000u) return (unsigned short)((u >> 16) | 0x0040u); // quiet NaN
  u += 0x7fffu + ((u >> 16) & 1u);   // round-to-nearest-even
  return (unsigned short)(u >> 16);
}

// ---------------------------------------------------------------------------
// Kernel 1: build w_eff = cos(ph)*w_real - sin(ph)*w_imag in bf16, stored as
// per-lane WMMA B fragments, slab-ordered so each (k, nQuad) slab of 32
// fragments (ib=0..7 x nb=0..3) is one contiguous 32KB block for the TDM.
// B layout used here, contiguous k matching A (16-bit, 16x16x32): lane = kHalf*16 + n; half j = K row
// (iBlk*32 + kHalf*16 + j), j = 0..15.
// ---------------------------------------------------------------------------
__global__ void weff_pack_kernel(const float* __restrict__ wr,
                                 const float* __restrict__ wi,
                                 const float* __restrict__ ph,
                                 unsigned short* __restrict__ wB) {
  int wave = (blockIdx.x * blockDim.x + threadIdx.x) >> 5;
  int lane = threadIdx.x & 31;
  if (wave >= KW * (CIN/32) * (COUT/16)) return;

  int k     = wave >> 7;        // / 128
  int rem   = wave & 127;
  int iBlk  = rem >> 4;
  int nTile = rem & 15;
  int nQuad = nTile >> 2;
  int nb    = nTile & 3;

  float c = cosf(ph[k]);
  float s = sinf(ph[k]);

  int n     = nTile * 16 + (lane & 15);
  int kHalf = lane >> 4;
  int iBase = iBlk * 32 + kHalf * 16;

  Pack16 p;
#pragma unroll
  for (int j = 0; j < 16; ++j) {
    int i = iBase + j;
    size_t idx = ((size_t)n * CIN + i) * KW + k;   // w[o][i][k]
    p.s[j] = f2bf(c * wr[idx] - s * wi[idx]);
  }
  size_t frag = (size_t)(((k * 4 + nQuad) * 8 + iBlk) * 4 + nb);
  unsigned short* dst = wB + frag * 512 + lane * 16;
  *(uint4*)(dst)     = p.q[0];
  *(uint4*)(dst + 8) = p.q[1];
}

// ---------------------------------------------------------------------------
// Kernel 2: convert x (f32, [B][S][CIN]) -> bf16 padded [B][S_PAD][CIN]
// grid = (1, S_PAD, BATCH), block = 128 (one uint32 = 2 bf16 per thread)
// ---------------------------------------------------------------------------
__global__ void x_pack_kernel(const float* __restrict__ x,
                              unsigned short* __restrict__ xp) {
  int t = blockIdx.y;
  int b = blockIdx.z;
  int i = threadIdx.x * 2;
  uint32_t o = 0u;
  if (t < S_LEN) {
    const float* p = x + ((size_t)b * S_LEN + t) * CIN + i;
    o = (uint32_t)f2bf(p[0]) | ((uint32_t)f2bf(p[1]) << 16);
  }
  *(uint32_t*)(xp + ((size_t)b * S_PAD + t) * CIN + i) = o;
}

// ---------------------------------------------------------------------------
// Kernel 3: implicit-GEMM conv via v_wmma_f32_16x16x32_bf16.
// Block = 256 threads = 8 waves; each wave owns a 16(M) x 64(N) tile.
// B slab (per k: 32KB) is DMA'd into LDS by the Tensor Data Mover
// (double-buffered) and read by all 8 waves via ds_load_b128.
// ---------------------------------------------------------------------------
__global__ __launch_bounds__(256) void conv_wmma_kernel(
    const unsigned short* __restrict__ xp,
    const unsigned short* __restrict__ wB,
    float* __restrict__ out) {
  __shared__ unsigned short ldsB[2][SLAB_HALFS];   // 2 x 32 KB

  const int lane   = threadIdx.x & 31;
  const int wv     = threadIdx.x >> 5;             // 0..7
  const int mTile  = blockIdx.x * 8 + wv;          // 0..4095
  const int nQuad  = blockIdx.y;                   // 0..3

  const int m0 = mTile * 16;
  const int b  = m0 >> 11;                         // / 2048
  const int t0 = m0 & 2047;

  const int laneM = lane & 15;
  const int c0    = (lane >> 4) * 16;
  const int mHi   = (lane >> 4) * 8;

  // base pointer for this lane's A row (row = t0 + laneM, plus k per step)
  const unsigned short* xbase =
      xp + ((size_t)b * S_PAD + t0 + laneM) * CIN + c0;

  // ---- B slab staging --------------------------------------------------
#if HAVE_TDM
  auto issue_slab = [&](int k, int buf) {
    if (wv == 0) {
      uint64_t gaddr = (uint64_t)(uintptr_t)wB +
                       ((uint64_t)(k * 4 + nQuad) * (uint64_t)SLAB_BYTES);
      uint32_t laddr = (uint32_t)(uintptr_t)&ldsB[buf][0];  // LDS byte offset
      v4u g0;
      g0[0] = 1u;                                  // count=1, user descriptor
      g0[1] = laddr;                               // lds_addr
      g0[2] = (uint32_t)gaddr;                     // global_addr[31:0]
      g0[3] = (uint32_t)((gaddr >> 32) & 0x01FFFFFFu) | (2u << 30); // type=2
      v8i g1;
      g1[0] = (int)(3u << 16);                     // data_size=8B, mask=0
      g1[1] = (int)(4096u << 16);                  // tensor_dim0 = 4096 elems
      g1[2] = (int)(1u << 16);                     // tensor_dim1 = 1
      g1[3] = (int)(4096u << 16);                  // tile_dim0 = 4096 elems
      g1[4] = 0;                                   // tile_dim1/2 unused
      g1[5] = 4096;                                // tensor_dim0_stride
      g1[6] = 0;
      g1[7] = 0;
      v4i z4 = {0, 0, 0, 0};
      v8i z8 = {0, 0, 0, 0, 0, 0, 0, 0};
      // 6-arg form (clang-23 / therock-10.0 headers)
      __builtin_amdgcn_tensor_load_to_lds(g0, g1, z4, z4, z8, 0);
    }
  };
  auto wait_slab = [&]() {
    if (wv == 0) __builtin_amdgcn_s_wait_tensorcnt(0);
  };
#else
  auto issue_slab = [&](int k, int buf) {
    const uint4* src = (const uint4*)(wB + (size_t)(k * 4 + nQuad) * SLAB_HALFS);
    uint4* dst = (uint4*)&ldsB[buf][0];
#pragma unroll
    for (int j = 0; j < 8; ++j)
      dst[threadIdx.x + j * 256] = src[threadIdx.x + j * 256];
  };
  auto wait_slab = [&]() {};
#endif

  v8f acc[4] = {v8f{}, v8f{}, v8f{}, v8f{}};

  // preload k = 0
  issue_slab(0, 0);
  wait_slab();
  __syncthreads();

  for (int k = 0; k < KW; ++k) {
    const int cur = k & 1;
    if (k + 1 < KW) issue_slab(k + 1, cur ^ 1);    // prefetch next slab

    const unsigned short* xk = xbase + (size_t)k * CIN;
    const uint4* slab = (const uint4*)&ldsB[cur][0];

#pragma unroll 2
    for (int ib = 0; ib < CIN / 32; ++ib) {
      FragA a;
      a.q[0] = *(const uint4*)(xk + ib * 32);        // K cols c0..c0+7
      a.q[1] = *(const uint4*)(xk + ib * 32 + 8);    // K cols c0+8..c0+15

#pragma unroll
      for (int nb = 0; nb < 4; ++nb) {
        const int fq = (ib * 4 + nb) * 64 + lane * 2; // uint4 index into slab
        FragB bb;
        bb.q[0] = slab[fq];
        bb.q[1] = slab[fq + 1];
        acc[nb] = __builtin_amdgcn_wmma_f32_16x16x32_bf16(
            /*neg_a=*/false, a.v, /*neg_b=*/false, bb.v,
            /*c_mod=*/(short)0, acc[nb],
            /*reuse_a=*/false, /*reuse_b=*/false);
      }
    }

    wait_slab();        // next slab fully in LDS (wave 0's TENSORcnt)
    __syncthreads();    // all waves done with cur, next buffer visible
  }

  // Store: C/D layout -> VGPR r is row M = mHi + r, col N = laneM.
#pragma unroll
  for (int nb = 0; nb < 4; ++nb) {
    const int ncol = (nQuad * 4 + nb) * 16 + laneM;
#pragma unroll
    for (int r = 0; r < 8; ++r) {
      const int t = t0 + mHi + r;
      if (t < T_OUT) {
        out[((size_t)b * T_OUT + t) * COUT + ncol] = acc[nb][r];
      }
    }
  }
}

// ---------------------------------------------------------------------------
extern "C" void kernel_launch(void* const* d_in, const int* in_sizes, int n_in,
                              void* d_out, int out_size, void* d_ws, size_t ws_size,
                              hipStream_t stream) {
  const float* x  = (const float*)d_in[0];
  const float* wr = (const float*)d_in[1];
  const float* wi = (const float*)d_in[2];
  const float* ph = (const float*)d_in[3];
  float* out = (float*)d_out;

  unsigned short* wB = (unsigned short*)d_ws;
  unsigned short* xp = (unsigned short*)((char*)d_ws + WB_BYTES);

  // 1) pack weights into slab-ordered WMMA B fragments (2048 waves)
  {
    dim3 grid(256), block(256);
    weff_pack_kernel<<<grid, block, 0, stream>>>(wr, wi, ph, wB);
  }
  // 2) convert + pad x to bf16
  {
    dim3 grid(1, S_PAD, BATCH), block(128);
    x_pack_kernel<<<grid, block, 0, stream>>>(x, xp);
  }
  // 3) WMMA implicit-GEMM conv with TDM-staged B
  {
    dim3 grid(512, 4), block(256);
    conv_wmma_kernel<<<grid, block, 0, stream>>>(xp, wB, out);
  }
}
